// RepresentationModel_82171314307531
// MI455X (gfx1250) — compile-verified, matrix operand order checked by reading, not run on hardware
//
#include <hip/hip_runtime.h>

// ---------------- Types for CDNA5 WMMA (gfx1250, wave32) ----------------
typedef __attribute__((ext_vector_type(16))) __bf16 v16bf;
typedef __attribute__((ext_vector_type(8)))  __bf16 v8bf;
typedef __attribute__((ext_vector_type(8)))  float  v8f;

// ---------------- Weight prep: fp32 [K,D] -> bf16 [D,K] ----------------
__global__ void k_wt_bf16t(const float* __restrict__ W, __bf16* __restrict__ Wt,
                           int K, int D) {
    int idx = blockIdx.x * blockDim.x + threadIdx.x;
    if (idx >= K * D) return;
    int d = idx / K;
    int k = idx - d * K;
    Wt[(size_t)d * K + k] = (__bf16)W[(size_t)k * D + d];
}

// ---------------- Generic split-A WMMA GEMM ----------------
// Out[M,D] = act( [A1 | A2][M, K1+K2] @ W[K,D] + bias ), W given as bf16 Wt[D,K].
// One wave computes a 16x64 output tile (4 WMMA accumulators sharing one A
// fragment). Block = 256 threads = 8 waves = 16x512 per block.
// Out-of-range rows/cols are CLAMPED on load (A row m only affects output row m,
// B col n only affects output col n) and guarded on store -> no divergent
// branches in the hot loop. act: 0=none, 1=relu, 2=tanh.
__global__ void k_gemm_wmma(const float* __restrict__ A1, int K1,
                            const float* __restrict__ A2, int K2,
                            const __bf16* __restrict__ Wt,
                            const float* __restrict__ bias,
                            float* __restrict__ Out,
                            int M, int D, int act) {
    const int K      = K1 + K2;
    const int lane   = threadIdx.x & 31;
    const int wid    = threadIdx.x >> 5;
    const int tile_m = blockIdx.x;
    const int ngrp   = blockIdx.y * 8 + wid;     // group of 4 n-tiles
    const int ntiles = (D + 15) >> 4;
    if (ngrp * 4 >= ntiles) return;              // wave-uniform

    const int mrow = lane & 15;
    const int koff = (lane < 16) ? 0 : 8;        // 16-bit A/B lane layout
    int mg = tile_m * 16 + mrow;
    if (mg >= M) mg = M - 1;                     // clamp (store is guarded)

    const __bf16* wrow[4];
#pragma unroll
    for (int j = 0; j < 4; ++j) {
        int nc = (ngrp * 4 + j) * 16 + (lane & 15);
        if (nc >= D) nc = D - 1;                 // clamp (store is guarded)
        wrow[j] = Wt + (size_t)nc * K;
    }

    v8f acc[4] = {{}, {}, {}, {}};

    // ---- main loops over K (A1 part, then optional A2 part) ----
    const float* arow1 = A1 + (size_t)mg * K1;
    for (int kk = 0; kk < K1; kk += 32) {
        const int kb = kk + koff;
        const float4 f0 = *(const float4*)(arow1 + kb);
        const float4 f1 = *(const float4*)(arow1 + kb + 4);
        const float4 f2 = *(const float4*)(arow1 + kb + 16);
        const float4 f3 = *(const float4*)(arow1 + kb + 20);
        v16bf a;
        a[0]  = (__bf16)f0.x; a[1]  = (__bf16)f0.y; a[2]  = (__bf16)f0.z; a[3]  = (__bf16)f0.w;
        a[4]  = (__bf16)f1.x; a[5]  = (__bf16)f1.y; a[6]  = (__bf16)f1.z; a[7]  = (__bf16)f1.w;
        a[8]  = (__bf16)f2.x; a[9]  = (__bf16)f2.y; a[10] = (__bf16)f2.z; a[11] = (__bf16)f2.w;
        a[12] = (__bf16)f3.x; a[13] = (__bf16)f3.y; a[14] = (__bf16)f3.z; a[15] = (__bf16)f3.w;
#pragma unroll
        for (int j = 0; j < 4; ++j) {
            const v8bf blo = *(const v8bf*)(wrow[j] + kk + koff);
            const v8bf bhi = *(const v8bf*)(wrow[j] + kk + koff + 16);
            const v16bf b = __builtin_shufflevector(blo, bhi,
                                0,1,2,3,4,5,6,7,8,9,10,11,12,13,14,15);
            acc[j] = __builtin_amdgcn_wmma_f32_16x16x32_bf16(
                         false, a, false, b, (short)0, acc[j], false, false);
        }
    }
    if (K2 > 0) {
        const float* arow2 = A2 + (size_t)mg * K2;
        for (int kk = 0; kk < K2; kk += 32) {
            const int kb = kk + koff;
            const float4 f0 = *(const float4*)(arow2 + kb);
            const float4 f1 = *(const float4*)(arow2 + kb + 4);
            const float4 f2 = *(const float4*)(arow2 + kb + 16);
            const float4 f3 = *(const float4*)(arow2 + kb + 20);
            v16bf a;
            a[0]  = (__bf16)f0.x; a[1]  = (__bf16)f0.y; a[2]  = (__bf16)f0.z; a[3]  = (__bf16)f0.w;
            a[4]  = (__bf16)f1.x; a[5]  = (__bf16)f1.y; a[6]  = (__bf16)f1.z; a[7]  = (__bf16)f1.w;
            a[8]  = (__bf16)f2.x; a[9]  = (__bf16)f2.y; a[10] = (__bf16)f2.z; a[11] = (__bf16)f2.w;
            a[12] = (__bf16)f3.x; a[13] = (__bf16)f3.y; a[14] = (__bf16)f3.z; a[15] = (__bf16)f3.w;
#pragma unroll
            for (int j = 0; j < 4; ++j) {
                const int kw = K1 + kk + koff;
                const v8bf blo = *(const v8bf*)(wrow[j] + kw);
                const v8bf bhi = *(const v8bf*)(wrow[j] + kw + 16);
                const v16bf b = __builtin_shufflevector(blo, bhi,
                                    0,1,2,3,4,5,6,7,8,9,10,11,12,13,14,15);
                acc[j] = __builtin_amdgcn_wmma_f32_16x16x32_bf16(
                             false, a, false, b, (short)0, acc[j], false, false);
            }
        }
    }

    // ---- Epilogue: bias + activation + bounds-checked store ----
    const int mbase = tile_m * 16 + ((lane < 16) ? 0 : 8);
#pragma unroll
    for (int j = 0; j < 4; ++j) {
        const int nc = (ngrp * 4 + j) * 16 + (lane & 15);
        if (nc >= D) continue;
        const float bv = bias ? bias[nc] : 0.f;
#pragma unroll
        for (int r = 0; r < 8; ++r) {
            int mo = mbase + r;
            if (mo < M) {
                float v = acc[j][r] + bv;
                if (act == 1)      v = fmaxf(v, 0.f);
                else if (act == 2) v = tanhf(v);
                Out[(size_t)mo * D + nc] = v;
            }
        }
    }
}

// ---------------- GCN aggregation: fixed +-1,+-2 + self stencil ----------------
static __device__ __forceinline__ float degf(int p, int Kb) {
    float dg = 1.f;                 // self loop
    if (p >= 1)       dg += 1.f;
    if (p >= 2)       dg += 1.f;
    if (p <= Kb - 2)  dg += 1.f;
    if (p <= Kb - 3)  dg += 1.f;
    return dg;
}

__global__ void k_gcn_agg(const float* __restrict__ Min, const float* __restrict__ bias,
                          float* __restrict__ Out, int N, int Kb, int D, int relu_flag) {
    const int vecD = D >> 2;
    size_t idx = (size_t)blockIdx.x * blockDim.x + threadIdx.x;
    if (idx >= (size_t)N * vecD) return;
    int node = (int)(idx / vecD);
    int dv   = (int)(idx - (size_t)node * vecD);
    int pos  = node % Kb;
    float di = rsqrtf(degf(pos, Kb));
    float ax = 0.f, ay = 0.f, az = 0.f, aw = 0.f;
#pragma unroll
    for (int d = -2; d <= 2; ++d) {
        int p = pos + d;
        if (p < 0 || p >= Kb) continue;
        float w = di * rsqrtf(degf(p, Kb));
        const float4 mv = *(const float4*)(Min + (size_t)(node + d) * D + dv * 4);
        ax += w * mv.x; ay += w * mv.y; az += w * mv.z; aw += w * mv.w;
    }
    const float4 b4 = *(const float4*)(bias + dv * 4);
    ax += b4.x; ay += b4.y; az += b4.z; aw += b4.w;
    if (relu_flag) {
        ax = fmaxf(ax, 0.f); ay = fmaxf(ay, 0.f);
        az = fmaxf(az, 0.f); aw = fmaxf(aw, 0.f);
    }
    *(float4*)(Out + (size_t)node * D + dv * 4) = make_float4(ax, ay, az, aw);
}

// ---------------- Attention scores: s[i] = tanh(u) . context (tanh pre-applied) ----------------
__global__ void k_scores(const float* __restrict__ U, const float* __restrict__ ctx,
                         float* __restrict__ S, int N, int D) {
    int lane = threadIdx.x & 31;
    int gw   = blockIdx.x * (blockDim.x >> 5) + (threadIdx.x >> 5);
    if (gw >= N) return;
    const float* row = U + (size_t)gw * D;
    float acc = 0.f;
    for (int d = lane; d < D; d += 32) acc += row[d] * ctx[d];
#pragma unroll
    for (int off = 16; off; off >>= 1) acc += __shfl_xor(acc, off, 32);
    if (lane == 0) S[gw] = acc;
}

// ---------------- Per-file segment softmax + attention pooling ----------------
__global__ void k_pool(const float* __restrict__ S, const float* __restrict__ Hblk,
                       float* __restrict__ Alpha, float* __restrict__ Hfile,
                       int Kb, int D) {
    const int f = blockIdx.x;
    const int t = threadIdx.x;            // blockDim.x == 128, Kb <= 128
    __shared__ float red[128];
    __shared__ float al[128];
    float s = (t < Kb) ? S[(size_t)f * Kb + t] : -3.4e38f;
    red[t] = s; __syncthreads();
#pragma unroll
    for (int off = 64; off; off >>= 1) {
        if (t < off) red[t] = fmaxf(red[t], red[t + off]);
        __syncthreads();
    }
    float mx = red[0]; __syncthreads();
    float ex = (t < Kb) ? expf(s - mx) : 0.f;
    red[t] = ex; __syncthreads();
#pragma unroll
    for (int off = 64; off; off >>= 1) {
        if (t < off) red[t] += red[t + off];
        __syncthreads();
    }
    float a = ex / red[0];
    al[t] = a;
    if (t < Kb) Alpha[(size_t)f * Kb + t] = a;
    __syncthreads();
    const float* base = Hblk + (size_t)f * Kb * D;
    for (int d = t; d < D; d += 128) {
        float acc = 0.f;
        for (int j = 0; j < Kb; ++j) acc += al[j] * base[(size_t)j * D + d];
        Hfile[(size_t)f * D + d] = acc;
    }
}

// ---------------- Bug-head: h_file @ W_bug + b_bug ----------------
__global__ void k_bug(const float* __restrict__ Hf, const float* __restrict__ Wb,
                      const float* __restrict__ bb, float* __restrict__ Out,
                      int F, int D) {
    int lane = threadIdx.x & 31;
    int gw   = blockIdx.x * (blockDim.x >> 5) + (threadIdx.x >> 5);
    if (gw >= F) return;
    const float* row = Hf + (size_t)gw * D;
    float acc = 0.f;
    for (int d = lane; d < D; d += 32) acc += row[d] * Wb[d];
#pragma unroll
    for (int off = 16; off; off >>= 1) acc += __shfl_xor(acc, off, 32);
    if (lane == 0) Out[gw] = acc + bb[0];
}

// ---------------- loss_ortho = mean((z_sh^T z_pr)^2) ----------------
__global__ void k_ortho(const float* __restrict__ Zs, const float* __restrict__ Zp,
                        int F, int DS, int DP, float* __restrict__ loss) {
    int idx = blockIdx.x * blockDim.x + threadIdx.x;
    float v = 0.f;
    if (idx < DS * DP) {
        int a = idx / DP, b = idx - (idx / DP) * DP;
        float g = 0.f;
        for (int f = 0; f < F; ++f)
            g += Zs[(size_t)f * DS + a] * Zp[(size_t)f * DP + b];
        v = g * g;
    }
    __shared__ float red[256];
    red[threadIdx.x] = v; __syncthreads();
#pragma unroll
    for (int off = 128; off; off >>= 1) {
        if (threadIdx.x < off) red[threadIdx.x] += red[threadIdx.x + off];
        __syncthreads();
    }
    if (threadIdx.x == 0) atomicAdd(loss, red[0] / (float)(DS * DP));
}

// ---------------- Host launcher ----------------
extern "C" void kernel_launch(void* const* d_in, const int* in_sizes, int n_in,
                              void* d_out, int out_size, void* d_ws, size_t ws_size,
                              hipStream_t stream) {
    const float* h_sem    = (const float*)d_in[0];
    const float* e_struct = (const float*)d_in[1];
    const float* W_fusion = (const float*)d_in[2];
    const float* b_fusion = (const float*)d_in[3];
    const float* W_g1     = (const float*)d_in[4];
    const float* b_g1     = (const float*)d_in[5];
    const float* W_g2     = (const float*)d_in[6];
    const float* b_g2     = (const float*)d_in[7];
    const float* W_att    = (const float*)d_in[8];
    const float* b_att    = (const float*)d_in[9];
    const float* context  = (const float*)d_in[10];
    const float* W_sh1    = (const float*)d_in[11];
    const float* b_sh1    = (const float*)d_in[12];
    const float* W_sh2    = (const float*)d_in[13];
    const float* b_sh2    = (const float*)d_in[14];
    const float* W_pr1    = (const float*)d_in[15];
    const float* b_pr1    = (const float*)d_in[16];
    const float* W_pr2    = (const float*)d_in[17];
    const float* b_pr2    = (const float*)d_in[18];
    const float* W_dpr    = (const float*)d_in[19];
    const float* b_dpr    = (const float*)d_in[20];
    const float* W_bug    = (const float*)d_in[21];
    const float* b_bug    = (const float*)d_in[22];

    const int DSEM = 768;
    const int N    = in_sizes[0] / DSEM;
    const int DSTR = in_sizes[1] / N;
    const int DH   = in_sizes[3];
    const int DSH  = in_sizes[12];
    const int DPR  = in_sizes[16];
    const int NP   = in_sizes[20];
    const int Kb   = 100;           // blocks per file (reference setup)
    const int F    = N / Kb;

    float* out = (float*)d_out;
    const size_t OFF_HBLK  = 0;
    const size_t OFF_ALPHA = OFF_HBLK + (size_t)N * DH;
    const size_t OFF_HFILE = OFF_ALPHA + (size_t)N;
    const size_t OFF_ZSH   = OFF_HFILE + (size_t)F * DH;
    const size_t OFF_ZPR   = OFF_ZSH + (size_t)F * DSH;
    const size_t OFF_LPD   = OFF_ZPR + (size_t)F * DPR;
    const size_t OFF_BUG   = OFF_LPD + (size_t)F * NP;
    const size_t OFF_LOSS  = OFF_BUG + (size_t)F;

    // Bump allocator over workspace
    char*  wsb = (char*)d_ws;
    size_t cur = 0;
    auto walloc = [&](size_t bytes) -> void* {
        void* p = wsb + cur;
        cur += (bytes + 255) & ~(size_t)255;
        return p;
    };
    float*  ws_a   = (float*)walloc((size_t)N * DH * 4);       // ping
    float*  ws_b   = (float*)walloc((size_t)N * DH * 4);       // pong
    float*  ws_sc  = (float*)walloc((size_t)N * 4);            // scores
    int     DHD    = (DSH > DPR) ? DSH : DPR;
    float*  ws_hid = (float*)walloc((size_t)F * DHD * 4);      // head hidden
    __bf16* wt_fus = (__bf16*)walloc((size_t)(DSEM + DSTR) * DH * 2);
    __bf16* wt_g1  = (__bf16*)walloc((size_t)DH * DH * 2);
    __bf16* wt_g2  = (__bf16*)walloc((size_t)DH * DH * 2);
    __bf16* wt_att = (__bf16*)walloc((size_t)DH * DH * 2);
    __bf16* wt_sh1 = (__bf16*)walloc((size_t)DH * DSH * 2);
    __bf16* wt_sh2 = (__bf16*)walloc((size_t)DSH * DSH * 2);
    __bf16* wt_pr1 = (__bf16*)walloc((size_t)DH * DPR * 2);
    __bf16* wt_pr2 = (__bf16*)walloc((size_t)DPR * DPR * 2);
    __bf16* wt_dpr = (__bf16*)walloc((size_t)DPR * NP * 2);

    auto prep = [&](const float* W, __bf16* Wt, int K, int D) {
        int tot = K * D;
        k_wt_bf16t<<<(tot + 255) / 256, 256, 0, stream>>>(W, Wt, K, D);
    };
    prep(W_fusion, wt_fus, DSEM + DSTR, DH);
    prep(W_g1,  wt_g1,  DH,  DH);
    prep(W_g2,  wt_g2,  DH,  DH);
    prep(W_att, wt_att, DH,  DH);
    prep(W_sh1, wt_sh1, DH,  DSH);
    prep(W_sh2, wt_sh2, DSH, DSH);
    prep(W_pr1, wt_pr1, DH,  DPR);
    prep(W_pr2, wt_pr2, DPR, DPR);
    prep(W_dpr, wt_dpr, DPR, NP);

    auto gemm = [&](const float* A1, int K1, const float* A2, int K2,
                    const __bf16* Wt, const float* bias, float* Out_,
                    int M, int Dd, int act) {
        int ntiles = (Dd + 15) / 16;          // 16-col tiles
        int ngrps  = (ntiles + 3) / 4;        // 4 tiles per wave
        dim3 grid((M + 15) / 16, (ngrps + 7) / 8);
        k_gemm_wmma<<<grid, 256, 0, stream>>>(A1, K1, A2, K2, Wt, bias, Out_, M, Dd, act);
    };
    auto agg = [&](const float* Min, const float* bias, float* Out_, int relu_flag) {
        size_t tot = (size_t)N * (DH / 4);
        k_gcn_agg<<<(unsigned)((tot + 255) / 256), 256, 0, stream>>>(
            Min, bias, Out_, N, Kb, DH, relu_flag);
    };

    // 1) h0 = [h_sem | e_struct] @ W_fusion + b_fusion        -> ws_a
    gemm(h_sem, DSEM, e_struct, DSTR, wt_fus, b_fusion, ws_a, N, DH, 0);
    // 2) t = h0 @ W_g1                                         -> ws_b
    gemm(ws_a, DH, nullptr, 0, wt_g1, nullptr, ws_b, N, DH, 0);
    // 3) h1 = relu(agg(t) + b_g1)                              -> ws_a
    agg(ws_b, b_g1, ws_a, 1);
    // 4) t = h1 @ W_g2                                         -> ws_b
    gemm(ws_a, DH, nullptr, 0, wt_g2, nullptr, ws_b, N, DH, 0);
    // 5) h_blk = agg(t) + b_g2                                 -> d_out
    agg(ws_b, b_g2, out + OFF_HBLK, 0);
    // 6) u = tanh(h_blk @ W_att + b_att)                       -> ws_b
    gemm(out + OFF_HBLK, DH, nullptr, 0, wt_att, b_att, ws_b, N, DH, 2);
    // 7) scores = u . context
    k_scores<<<(N + 7) / 8, 256, 0, stream>>>(ws_b, context, ws_sc, N, DH);
    // 8) segment softmax + pooling -> alpha, h_file
    k_pool<<<F, 128, 0, stream>>>(ws_sc, out + OFF_HBLK,
                                  out + OFF_ALPHA, out + OFF_HFILE, Kb, DH);
    // 9) shared head
    gemm(out + OFF_HFILE, DH, nullptr, 0, wt_sh1, b_sh1, ws_hid, F, DSH, 1);
    gemm(ws_hid, DSH, nullptr, 0, wt_sh2, b_sh2, out + OFF_ZSH, F, DSH, 0);
    // 10) project head
    gemm(out + OFF_HFILE, DH, nullptr, 0, wt_pr1, b_pr1, ws_hid, F, DPR, 1);
    gemm(ws_hid, DPR, nullptr, 0, wt_pr2, b_pr2, out + OFF_ZPR, F, DPR, 0);
    // 11) project-domain logits
    gemm(out + OFF_ZPR, DPR, nullptr, 0, wt_dpr, b_dpr, out + OFF_LPD, F, NP, 0);
    // 12) bug logit
    k_bug<<<(F + 7) / 8, 256, 0, stream>>>(out + OFF_HFILE, W_bug, b_bug,
                                           out + OFF_BUG, F, DH);
    // 13) ortho loss
    hipMemsetAsync(out + OFF_LOSS, 0, sizeof(float), stream);
    int gtot = DSH * DPR;
    k_ortho<<<(gtot + 255) / 256, 256, 0, stream>>>(out + OFF_ZSH, out + OFF_ZPR,
                                                    F, DSH, DPR, out + OFF_LOSS);
    (void)n_in; (void)out_size; (void)ws_size;
}
